// QuantLinear_6158983102529
// MI455X (gfx1250) — matrix operand order repaired and checked by reading, not compiled, not run on hardware
//
#include <hip/hip_runtime.h>

typedef __attribute__((ext_vector_type(16))) _Float16 v16h;
typedef __attribute__((ext_vector_type(8)))  _Float16 v8h;
typedef __attribute__((ext_vector_type(2)))  _Float16 v2h;
typedef __attribute__((ext_vector_type(8)))  float    v8f;
typedef __attribute__((ext_vector_type(4)))  int      v4i;

#ifndef __has_builtin
#define __has_builtin(x) 0
#endif

#if __has_builtin(__builtin_amdgcn_global_load_async_to_lds_b128) && \
    __has_builtin(__builtin_amdgcn_s_wait_asynccnt)
#define HAVE_ASYNC 1
#else
#define HAVE_ASYNC 0
#endif

#define Mdim 8192
#define Kdim 4096
#define Ndim 4096

#define BM 128
#define BN 128
#define BK 64
#define LSTRIDE 72                  // halves per LDS row (64 + 8 pad, avoids bank conflicts)
#define STAGE (BM * LSTRIDE)        // halves per double-buffer stage

#define MT 4                        // 16-row M tiles per wave
#define NT 2                        // 16-col N tiles per wave

// Pointer types matching the async builtin's signature: v4i in as(1)/as(3).
typedef __attribute__((address_space(1))) v4i* gv4i_ptr;
typedef __attribute__((address_space(3))) v4i* lv4i_ptr;

union Bfrag { v16h v; v2h p[8]; };
union Afrag { v16h v; v8h h[2]; };

// Async-copy one BMxBK fp16 tile of A into LDS. 1024 16B chunks, 256 threads x 4.
__device__ __forceinline__ void stage_async(const _Float16* __restrict__ A,
                                            _Float16* lbuf, int tid,
                                            int block_m, int k0) {
#if HAVE_ASYNC
#pragma unroll
  for (int t = 0; t < 4; ++t) {
    const int chunk = tid + t * 256;
    const int r  = chunk >> 3;
    const int c8 = (chunk & 7) << 3;
    const _Float16* g = A + (size_t)(block_m + r) * Kdim + k0 + c8;
    _Float16* l = lbuf + r * LSTRIDE + c8;
    __builtin_amdgcn_global_load_async_to_lds_b128(
        (gv4i_ptr)(unsigned long long)g,
        (lv4i_ptr)(unsigned)(unsigned long long)l,
        0, 0);
  }
#else
  (void)A; (void)lbuf; (void)tid; (void)block_m; (void)k0;
#endif
}

// Fast int4 -> fp16 dequant, 8 nibbles of q -> out[0..3] (v2h each).
// 0x6400|nib is fp16 (1024+nib); fma with (s, -1032*s) gives (nib-8)*s.
__device__ __forceinline__ void dq8(unsigned q, v2h s2, v2h c2, v2h* out) {
  const unsigned MAGIC = 0x64006400u;
  const unsigned t0 = ( q        & 0x0000000Fu) | ((q << 12) & 0x000F0000u) | MAGIC;
  const unsigned t1 = ((q >>  8) & 0x0000000Fu) | ((q <<  4) & 0x000F0000u) | MAGIC;
  const unsigned t2 = ((q >> 16) & 0x0000000Fu) | ((q >>  4) & 0x000F0000u) | MAGIC;
  const unsigned t3 = ((q >> 24) & 0x0000000Fu) | ((q >> 12) & 0x000F0000u) | MAGIC;
  out[0] = __builtin_bit_cast(v2h, t0) * s2 + c2;   // v_pk_fma_f16
  out[1] = __builtin_bit_cast(v2h, t1) * s2 + c2;
  out[2] = __builtin_bit_cast(v2h, t2) * s2 + c2;
  out[3] = __builtin_bit_cast(v2h, t3) * s2 + c2;
}

__global__ __launch_bounds__(256, 1)
void wmma_w4a16_gemm(const _Float16* __restrict__ A,
                     const int* __restrict__ qweight,
                     const _Float16* __restrict__ scales,
                     const _Float16* __restrict__ bias,
                     _Float16* __restrict__ C)
{
  __shared__ _Float16 lds_a[2 * STAGE];

  const int tid  = threadIdx.x;
  const int lane = tid & 31;
  const int wave = tid >> 5;
  const int wm = wave & 1;    // 0..1 : 64-row slab
  const int wn = wave >> 1;   // 0..3 : 32-col slab

  const int block_n = blockIdx.x * BN;
  const int block_m = blockIdx.y * BM;

  const int ln = lane & 15;   // column within 16x16 tile (B/C/D), row for A frag
  const int lk = lane >> 4;   // half-wave selector

  v8f acc[MT][NT];
#pragma unroll
  for (int mt = 0; mt < MT; ++mt)
#pragma unroll
    for (int nt = 0; nt < NT; ++nt)
#pragma unroll
      for (int r = 0; r < 8; ++r) acc[mt][nt][r] = 0.0f;

  const int KITERS = Kdim / BK;   // 64

#if HAVE_ASYNC
  stage_async(A, lds_a, tid, block_m, 0);
#endif

  for (int kit = 0; kit < KITERS; ++kit) {
    const int k0 = kit * BK;

#if HAVE_ASYNC
    _Float16* cur = lds_a + (kit & 1) * STAGE;
    __builtin_amdgcn_s_wait_asynccnt(0);   // my stage-kit loads landed in LDS
    __syncthreads();                       // everyone's landed + prev compute done
    if (kit + 1 < KITERS)
      stage_async(A, lds_a + ((kit + 1) & 1) * STAGE, tid, block_m, k0 + BK);
#else
    _Float16* cur = lds_a;
    v8h stg[4];
#pragma unroll
    for (int t = 0; t < 4; ++t) {
      const int chunk = tid + t * 256;
      const int r  = chunk >> 3;
      const int c8 = (chunk & 7) << 3;
      stg[t] = *(const v8h*)(A + (size_t)(block_m + r) * Kdim + k0 + c8);
    }
    __syncthreads();                       // prev compute done, safe to overwrite
#pragma unroll
    for (int t = 0; t < 4; ++t) {
      const int chunk = tid + t * 256;
      const int r  = chunk >> 3;
      const int c8 = (chunk & 7) << 3;
      *(v8h*)(cur + r * LSTRIDE + c8) = stg[t];
    }
    __syncthreads();                       // tile visible to all waves
#endif

    // Per-group scale + folded zero-point constant, per N tile (splat v2h).
    const int grp = k0 >> 7;
    v2h s2[NT], c2[NT];
#pragma unroll
    for (int nt = 0; nt < NT; ++nt) {
      const int n = block_n + wn * (NT * 16) + nt * 16 + ln;
      const _Float16 s = scales[(size_t)grp * Ndim + n];
      const _Float16 c = s * (_Float16)(-1032.0f);
      s2[nt][0] = s; s2[nt][1] = s;
      c2[nt][0] = c; c2[nt][1] = c;
    }

#pragma unroll
    for (int ks = 0; ks < 2; ++ks) {
      const int kk = ks * 32;

      // A fragments from LDS, contiguous k order (not the ISA table order):
      // lanes 0-15: K 0-15 ; lanes 16-31: K 16-31 (contiguous, matches B)
      Afrag af[MT];
#pragma unroll
      for (int mt = 0; mt < MT; ++mt) {
        const _Float16* p =
            cur + (wm * (MT * 16) + mt * 16 + ln) * LSTRIDE + kk + lk * 16;
        af[mt].h[0] = *(const v8h*)p;         // ds_load_b128
        af[mt].h[1] = *(const v8h*)(p + 8);   // ds_load_b128
      }

#pragma unroll
      for (int nt = 0; nt < NT; ++nt) {
        // B fragment: lane holds column n, 16 consecutive K nibbles = 2 int32s.
        const int n    = block_n + wn * (NT * 16) + nt * 16 + ln;
        const int kloc = k0 + kk + lk * 16;
        const int qrow = kloc >> 3;
        const unsigned q0 = (unsigned)qweight[(size_t)qrow * Ndim + n];
        const unsigned q1 = (unsigned)qweight[(size_t)(qrow + 1) * Ndim + n];

        Bfrag bf;
        dq8(q0, s2[nt], c2[nt], &bf.p[0]);   // K slots 0..7
        dq8(q1, s2[nt], c2[nt], &bf.p[4]);   // K slots 8..15

        // One dequant feeds 4 WMMAs.
#pragma unroll
        for (int mt = 0; mt < MT; ++mt) {
          acc[mt][nt] = __builtin_amdgcn_wmma_f32_16x16x32_f16(
              false, af[mt].v, false, bf.v, (short)0, acc[mt][nt], false, false);
        }
      }
    }
  }

  // Epilogue: fp32 acc + bias -> fp16, per C/D layout (VGPR r <-> M=r / M=r+8)
#pragma unroll
  for (int nt = 0; nt < NT; ++nt) {
    const int n = block_n + wn * (NT * 16) + nt * 16 + ln;
    const float bv = (float)bias[n];
#pragma unroll
    for (int mt = 0; mt < MT; ++mt) {
      const int row0 = block_m + wm * (MT * 16) + mt * 16 + lk * 8;
#pragma unroll
      for (int r = 0; r < 8; ++r) {
        C[(size_t)(row0 + r) * Ndim + n] = (_Float16)(acc[mt][nt][r] + bv);
      }
    }
  }
}

extern "C" void kernel_launch(void* const* d_in, const int* in_sizes, int n_in,
                              void* d_out, int out_size, void* d_ws, size_t ws_size,
                              hipStream_t stream) {
  (void)in_sizes; (void)n_in; (void)out_size; (void)d_ws; (void)ws_size;
  const _Float16* A       = (const _Float16*)d_in[0];
  const int*      qweight = (const int*)d_in[1];
  const _Float16* scales  = (const _Float16*)d_in[2];
  const _Float16* bias    = (const _Float16*)d_in[3];
  _Float16*       Cout    = (_Float16*)d_out;

  dim3 grid(Ndim / BN, Mdim / BM);   // 32 x 64 blocks
  wmma_w4a16_gemm<<<grid, 256, 0, stream>>>(A, qweight, scales, bias, Cout);
}
